// Fs_net_37400575213721
// MI455X (gfx1250) — matrix operand order repaired and checked by reading, not compiled, not run on hardware
//
#include <hip/hip_runtime.h>
#include <hip/hip_bf16.h>
#include <math.h>

// ---------------------------------------------------------------------------
// Seq2seq bi-GRU stack for gfx1250 (MI455X), all matmuls via
// v_wmma_f32_16x16x32_f16 (f16 operands, f32 accumulate).
//
// Key transforms vs reference:
//  * enc-L1 input projection = table lookup (emb@W+b0 precomputed, VOCAB=128)
//  * dec-L1 input projection = one GEMM (decoder input constant over t)
//  * layer-2 cells: fused per-step GEMM [32,384]x[384,512] with stacked
//    zero-padded weights producing [z | r | h@U_h | x@W_h] so the
//    reset-after GRU candidate tanh(x_h + r*rec_h) is exact.
//  * scan_small holds all 12 loop-invariant B fragments in registers.
//  * scan_big software-prefetches next step's inputs during the gate phase.
//  * sRec is column-major so D-fragment spills are single ds_store_b128s.
// ---------------------------------------------------------------------------

typedef __attribute__((ext_vector_type(16))) _Float16 v16h;
typedef __attribute__((ext_vector_type(8)))  _Float16 v8h;
typedef __attribute__((ext_vector_type(8)))  float    v8f;
typedef _Float16 h2 __attribute__((ext_vector_type(2)));

#define B_   512
#define T_   256
#define BT   32            // batch rows per scan block
#define NBLK (B_ / BT)     // 16 blocks per direction

struct ScanP {
    const _Float16* frag;      // packed f16 WMMA-B weight fragments
    const float*    b0;        // input bias (fused layers only)
    const float*    b1;        // recurrent bias
    const float*    xp;        // xproj table [128,384] (tok mode) or [B,384]
    const int*      tok;       // X [B,T] or nullptr (const-xp mode)
    float*          out_hs;    // per-step hidden output (or nullptr)
    long long       ohs_sb;    // out_hs stride per batch row
    long long       ohs_st;    // out_hs stride per time step
    float*          out_state; // final state [B, pitch] (or nullptr)
    int             os_pitch;
    int             reverse;   // 0 = forward scan, 1 = backward scan
};

__device__ __forceinline__ v8f wmma16(v16h a, v16h b, v8f c) {
    return __builtin_amdgcn_wmma_f32_16x16x32_f16(
        false, a, false, b, (short)0, c, false, false);
}

// Build A fragment (16x32 f16, contiguous k = 16*hi + s, same as the B packing) from LDS row-major f16 [BT, KD].
__device__ __forceinline__ v16h buildA(const _Float16* sA, int KD,
                                       int mt, int kt, int r16, int hi) {
    v16h a;
    const _Float16* rowp = sA + (mt * 16 + r16) * KD + kt * 32 + hi * 16;
#pragma unroll
    for (int jj = 0; jj < 8; ++jj) {
        int k = 2 * jj;                             // K within lane-half
        h2 p = *(const h2*)(rowp + k);
        a[2 * jj]     = p.x;
        a[2 * jj + 1] = p.y;
    }
    return a;
}

// Store D fragment (16x16 f32) into column-major LDS f16 [ND][BT]:
// the 8 per-lane elements are M-consecutive at fixed N -> one b128 store.
__device__ __forceinline__ void storeAccCM(_Float16* sRec,
                                           int mt, int nt, int r16, int hi, v8f acc) {
    v8h p;
#pragma unroll
    for (int i = 0; i < 8; ++i) p[i] = (_Float16)acc[i];
    *(v8h*)(sRec + (nt * 16 + r16) * BT + mt * 16 + hi * 8) = p;
}

__device__ __forceinline__ float sigm(float x) { return 1.f / (1.f + __expf(-x)); }

// ---------------------------------------------------------------------------
// scan_small: layer-1 GRU cells. rec = h@U (K=128, N=384); xproj gathered
// (embedding-table mode) or constant-per-row mode. Always stores hs.
// B fragments (12 per wave) are loop-invariant -> held in registers.
// ---------------------------------------------------------------------------
__global__ __launch_bounds__(256) void k_scan_small(ScanP Pf, ScanP Pb) {
    ScanP P = blockIdx.y ? Pb : Pf;
    __shared__ _Float16 sA[BT * 128];    // h state (f16), row-major
    __shared__ _Float16 sRec[384 * BT];  // GEMM result, column-major
    const int tid  = threadIdx.x;
    const int lane = tid & 31, wave = tid >> 5;
    const int r16  = lane & 15, hi = lane >> 4;
    const int gb0  = blockIdx.x * BT;
    const int nt0  = wave * 3;           // 8 waves x 3 n-tiles = 24

    // Hoist loop-invariant weight fragments into registers (12 x v16h).
    v16h Breg[12];
#pragma unroll
    for (int kt = 0; kt < 4; ++kt)
#pragma unroll
        for (int i = 0; i < 3; ++i)
            Breg[kt * 3 + i] =
                *(const v16h*)(P.frag + (((kt * 24 + nt0 + i) * 32 + lane) * 16));

    for (int i = tid; i < BT * 128; i += 256) sA[i] = (_Float16)0.f;
    __syncthreads();

    for (int s = 0; s < T_; ++s) {
        const int t = P.reverse ? (T_ - 1 - s) : s;
        // GEMM: sRec[384][BT] = (h @ U)^T   (B operands from registers)
        for (int mt = 0; mt < BT / 16; ++mt) {
            v8f a0 = {}, a1 = {}, a2 = {};
#pragma unroll
            for (int kt = 0; kt < 4; ++kt) {
                v16h A = buildA(sA, 128, mt, kt, r16, hi);
                a0 = wmma16(A, Breg[kt * 3 + 0], a0);
                a1 = wmma16(A, Breg[kt * 3 + 1], a1);
                a2 = wmma16(A, Breg[kt * 3 + 2], a2);
            }
            storeAccCM(sRec, mt, nt0 + 0, r16, hi, a0);
            storeAccCM(sRec, mt, nt0 + 1, r16, hi, a1);
            storeAccCM(sRec, mt, nt0 + 2, r16, hi, a2);
        }
        __syncthreads();
        // Gates
#pragma unroll 4
        for (int e = tid; e < BT * 128; e += 256) {
            int row = e >> 7, c = e & 127;
            int gb = gb0 + row;
            const float* xr = P.tok ? (P.xp + (long long)P.tok[gb * T_ + t] * 384)
                                    : (P.xp + (long long)gb * 384);
            float xpz = xr[c], xpr = xr[128 + c], xph = xr[256 + c];
            float recz = (float)sRec[(c)       * BT + row] + P.b1[c];
            float recr = (float)sRec[(128 + c) * BT + row] + P.b1[128 + c];
            float rech = (float)sRec[(256 + c) * BT + row] + P.b1[256 + c];
            float z  = sigm(xpz + recz);
            float r  = sigm(xpr + recr);
            float hh = tanhf(xph + r * rech);
            float hold = (float)sA[row * 128 + c];
            float hn = z * hold + (1.f - z) * hh;
            sA[row * 128 + c] = (_Float16)hn;
            P.out_hs[(long long)gb * P.ohs_sb + (long long)t * P.ohs_st + c] = hn;
            if (P.out_state && s == T_ - 1)
                P.out_state[gb * P.os_pitch + c] = hn;
        }
        __syncthreads();
    }
}

// ---------------------------------------------------------------------------
// scan_big: layer-2 GRU cells. Fused GEMM [BT,384]x[384,512] producing
// columns [z | r | rec_h | x_h]. Inputs are previous layer's fw/bw hs.
// Next step's inputs are prefetched into registers during the gate phase.
// ---------------------------------------------------------------------------
__global__ __launch_bounds__(256) void k_scan_big(ScanP Pf, ScanP Pb,
                                                  const float* hsfw,
                                                  const float* hsbw) {
    ScanP P = blockIdx.y ? Pb : Pf;
    __shared__ _Float16 sA[BT * 384];    // [h | xfw | xbw] (f16), row-major
    __shared__ _Float16 sRec[512 * BT];  // GEMM result, column-major
    const int tid  = threadIdx.x;
    const int lane = tid & 31, wave = tid >> 5;
    const int r16  = lane & 15, hi = lane >> 4;
    const int gb0  = blockIdx.x * BT;
    const int nt0  = wave * 4;           // 8 waves x 4 n-tiles = 32

    // init h = 0 and stage t=0 inputs
    {
        const int t0 = P.reverse ? (T_ - 1) : 0;
#pragma unroll 4
        for (int e = tid; e < BT * 128; e += 256) {
            int row = e >> 7, c = e & 127;
            long long src = ((long long)t0 * B_ + (gb0 + row)) * 128 + c;
            sA[row * 384 + c]       = (_Float16)0.f;
            sA[row * 384 + 128 + c] = (_Float16)hsfw[src];
            sA[row * 384 + 256 + c] = (_Float16)hsbw[src];
        }
    }
    __syncthreads();

    for (int s = 0; s < T_; ++s) {
        const int t  = P.reverse ? (T_ - 1 - s) : s;
        const int sn = (s + 1 < T_) ? s + 1 : s;       // clamped next step
        const int tn = P.reverse ? (T_ - 1 - sn) : sn;
        // GEMM: sRec[512][BT] = ([h|xfw|xbw] @ S)^T
        for (int mt = 0; mt < BT / 16; ++mt) {
            v8f a0 = {}, a1 = {}, a2 = {}, a3 = {};
#pragma unroll
            for (int kt = 0; kt < 12; ++kt) {
                v16h A  = buildA(sA, 384, mt, kt, r16, hi);
                v16h B0 = *(const v16h*)(P.frag + (((kt * 32 + nt0 + 0) * 32 + lane) * 16));
                v16h B1 = *(const v16h*)(P.frag + (((kt * 32 + nt0 + 1) * 32 + lane) * 16));
                v16h B2 = *(const v16h*)(P.frag + (((kt * 32 + nt0 + 2) * 32 + lane) * 16));
                v16h B3 = *(const v16h*)(P.frag + (((kt * 32 + nt0 + 3) * 32 + lane) * 16));
                a0 = wmma16(A, B0, a0);
                a1 = wmma16(A, B1, a1);
                a2 = wmma16(A, B2, a2);
                a3 = wmma16(A, B3, a3);
            }
            storeAccCM(sRec, mt, nt0 + 0, r16, hi, a0);
            storeAccCM(sRec, mt, nt0 + 1, r16, hi, a1);
            storeAccCM(sRec, mt, nt0 + 2, r16, hi, a2);
            storeAccCM(sRec, mt, nt0 + 3, r16, hi, a3);
        }
        // Prefetch next step's inputs (independent of h / this step's gates).
        float pf[16], pb[16];
#pragma unroll
        for (int i = 0; i < 16; ++i) {
            int e = tid + i * 256;
            int row = e >> 7, c = e & 127;
            long long src = ((long long)tn * B_ + (gb0 + row)) * 128 + c;
            pf[i] = hsfw[src];
            pb[i] = hsbw[src];
        }
        __syncthreads();
        // Gates: z | r | rec_h | x_h   (reads sRec + sA h-cols only)
#pragma unroll 4
        for (int i = 0; i < 16; ++i) {
            int e = tid + i * 256;
            int row = e >> 7, c = e & 127;
            int gb = gb0 + row;
            float zp   = (float)sRec[(c)       * BT + row] + P.b0[c]       + P.b1[c];
            float rp   = (float)sRec[(128 + c) * BT + row] + P.b0[128 + c] + P.b1[128 + c];
            float rech = (float)sRec[(256 + c) * BT + row] + P.b1[256 + c];
            float xh   = (float)sRec[(384 + c) * BT + row] + P.b0[256 + c];
            float z  = sigm(zp);
            float r  = sigm(rp);
            float hh = tanhf(xh + r * rech);
            float hold = (float)sA[row * 384 + c];
            float hn = z * hold + (1.f - z) * hh;
            sA[row * 384 + c] = (_Float16)hn;
            if (P.out_hs)
                P.out_hs[(long long)gb * P.ohs_sb + (long long)t * P.ohs_st + c] = hn;
            if (P.out_state && s == T_ - 1)
                P.out_state[gb * P.os_pitch + c] = hn;
        }
        // Commit prefetched inputs for the next step (x-cols; disjoint from
        // gate reads/writes, and this step's GEMM reads finished pre-barrier).
#pragma unroll
        for (int i = 0; i < 16; ++i) {
            int e = tid + i * 256;
            int row = e >> 7, c = e & 127;
            sA[row * 384 + 128 + c] = (_Float16)pf[i];
            sA[row * 384 + 256 + c] = (_Float16)pb[i];
        }
        __syncthreads();
    }
}

// ---------------------------------------------------------------------------
// Weight packing into WMMA B-fragment layout (32x16 K-tiles, f16):
//   frag[((kt*NT + nt)*32 + lane)*16 + j] = S[kt*32 + j + (lane>>4)*16][nt*16 + (lane&15)]
// ---------------------------------------------------------------------------
__global__ void k_pack_small(const float* U, _Float16* frag) {
    int id = blockIdx.x * 256 + threadIdx.x;
    if (id >= 4 * 24 * 32) return;
    int kt = id / (24 * 32), rem = id % (24 * 32);
    int nt = rem / 32, lane = rem % 32;
    int r16 = lane & 15, hi = lane >> 4;
    _Float16* o = frag + ((long long)id * 16);
#pragma unroll
    for (int j = 0; j < 16; ++j) {
        int k = kt * 32 + j + hi * 16;
        o[j] = (_Float16)U[k * 384 + nt * 16 + r16];
    }
}

__global__ void k_pack_big(const float* U, const float* W, _Float16* frag) {
    int id = blockIdx.x * 256 + threadIdx.x;
    if (id >= 12 * 32 * 32) return;
    int kt = id / (32 * 32), rem = id % (32 * 32);
    int nt = rem / 32, lane = rem % 32;
    int r16 = lane & 15, hi = lane >> 4;
    _Float16* o = frag + ((long long)id * 16);
#pragma unroll
    for (int j = 0; j < 16; ++j) {
        int k = kt * 32 + j + hi * 16;   // 0..383
        int n = nt * 16 + r16;           // 0..511
        float v;
        if (n < 256)      v = (k < 128) ? U[k * 384 + n] : W[(k - 128) * 384 + n];
        else if (n < 384) v = (k < 128) ? U[k * 384 + n] : 0.f;
        else              v = (k < 128) ? 0.f : W[(k - 128) * 384 + (n - 128)];
        o[j] = (_Float16)v;
    }
}

// out[M,384] = A[M,K] @ Wt[K,384] + b0   (small projections, VALU is fine)
__global__ void k_proj_addb(const float* A, const float* Wt, const float* b0,
                            float* out, int M, int K) {
    int id = blockIdx.x * 256 + threadIdx.x;
    if (id >= M * 384) return;
    int m = id / 384, n = id % 384;
    float acc = b0[n];
    for (int k = 0; k < K; ++k) acc += A[m * K + k] * Wt[k * 384 + n];
    out[id] = acc;
}

// Classifier head: cls[1024] -> selu dense 128 -> sigmoid dense 25
__global__ __launch_bounds__(128) void k_head(const float* ef, const float* df,
                                              const float* W1, const float* b1,
                                              const float* W2, const float* b2,
                                              float* out) {
    __shared__ float cls[1024];
    __shared__ float d1[128];
    int b = blockIdx.x, tid = threadIdx.x;
    for (int i = tid; i < 1024; i += 128) {
        int seg = i >> 8, idx = i & 255;
        float e = ef[b * 256 + idx], d = df[b * 256 + idx];
        cls[i] = (seg == 0) ? e : (seg == 1) ? d : (seg == 2) ? e * d : fabsf(e - d);
    }
    __syncthreads();
    float acc = b1[tid];
    for (int k = 0; k < 1024; ++k) acc += cls[k] * W1[k * 128 + tid];
    const float scale = 1.0507009873554805f, alpha = 1.6732632423543772f;
    d1[tid] = acc > 0.f ? scale * acc : scale * alpha * (__expf(acc) - 1.f);
    __syncthreads();
    if (tid < 25) {
        float a2 = b2[tid];
        for (int k = 0; k < 128; ++k) a2 += d1[k] * W2[k * 25 + tid];
        out[b * 25 + tid] = 1.f / (1.f + __expf(-a2));
    }
}

// ---------------------------------------------------------------------------
// Input order assumes jax-style tree flattening (dict keys sorted):
// 0:X 1:W1 2:W2 3:b1 4:b2
// 5-16: dec[0].bw{U,W,b} dec[0].fw{U,W,b} dec[1].bw{U,W,b} dec[1].fw{U,W,b}
// 17: emb
// 18-29: enc[0].bw{U,W,b} enc[0].fw{U,W,b} enc[1].bw{U,W,b} enc[1].fw{U,W,b}
// ---------------------------------------------------------------------------
extern "C" void kernel_launch(void* const* d_in, const int* in_sizes, int n_in,
                              void* d_out, int out_size, void* d_ws, size_t ws_size,
                              hipStream_t stream) {
    (void)in_sizes; (void)n_in; (void)out_size; (void)ws_size;
    const int*   X    = (const int*)  d_in[0];
    const float* W1   = (const float*)d_in[1];
    const float* W2   = (const float*)d_in[2];
    const float* b1v  = (const float*)d_in[3];
    const float* b2v  = (const float*)d_in[4];
    const float* d0bU = (const float*)d_in[5];
    const float* d0bW = (const float*)d_in[6];
    const float* d0bb = (const float*)d_in[7];
    const float* d0fU = (const float*)d_in[8];
    const float* d0fW = (const float*)d_in[9];
    const float* d0fb = (const float*)d_in[10];
    const float* d1bU = (const float*)d_in[11];
    const float* d1bW = (const float*)d_in[12];
    const float* d1bb = (const float*)d_in[13];
    const float* d1fU = (const float*)d_in[14];
    const float* d1fW = (const float*)d_in[15];
    const float* d1fb = (const float*)d_in[16];
    const float* emb  = (const float*)d_in[17];
    const float* e0bU = (const float*)d_in[18];
    const float* e0bW = (const float*)d_in[19];
    const float* e0bb = (const float*)d_in[20];
    const float* e0fU = (const float*)d_in[21];
    const float* e0fW = (const float*)d_in[22];
    const float* e0fb = (const float*)d_in[23];
    const float* e1bU = (const float*)d_in[24];
    const float* e1bW = (const float*)d_in[25];
    const float* e1bb = (const float*)d_in[26];
    const float* e1fU = (const float*)d_in[27];
    const float* e1fW = (const float*)d_in[28];
    const float* e1fb = (const float*)d_in[29];

    // Workspace layout
    float* F     = (float*)d_ws;
    float* hsfw  = F;                        // [T,B,128]
    float* hsbw  = hsfw + (size_t)T_ * B_ * 128;
    float* embWf = hsbw + (size_t)T_ * B_ * 128;   // [128,384]
    float* embWb = embWf + 128 * 384;
    float* encF  = embWb + 128 * 384;        // [B,256] (fw|bw)
    float* decF  = encF + B_ * 256;
    float* xpDf  = decF + B_ * 256;          // [B,384]
    float* xpDb  = xpDf + B_ * 384;
    _Float16* Hb  = (_Float16*)(xpDb + B_ * 384);
    _Float16* fE1f = Hb;                      // small frags: 4*24*32*16 halfs
    _Float16* fE1b = fE1f + 49152;
    _Float16* fD1f = fE1b + 49152;
    _Float16* fD1b = fD1f + 49152;
    _Float16* fE2f = fD1b + 49152;            // big frags: 12*32*32*16 halfs
    _Float16* fE2b = fE2f + 196608;
    _Float16* fD2f = fE2b + 196608;
    _Float16* fD2b = fD2f + 196608;

    // 1) embedding xproj tables + weight packing
    k_proj_addb<<<(128 * 384 + 255) / 256, 256, 0, stream>>>(emb, e0fW, e0fb, embWf, 128, 128);
    k_proj_addb<<<(128 * 384 + 255) / 256, 256, 0, stream>>>(emb, e0bW, e0bb, embWb, 128, 128);
    k_pack_small<<<(3072 + 255) / 256, 256, 0, stream>>>(e0fU, fE1f);
    k_pack_small<<<(3072 + 255) / 256, 256, 0, stream>>>(e0bU, fE1b);
    k_pack_small<<<(3072 + 255) / 256, 256, 0, stream>>>(d0fU, fD1f);
    k_pack_small<<<(3072 + 255) / 256, 256, 0, stream>>>(d0bU, fD1b);
    k_pack_big<<<(12288 + 255) / 256, 256, 0, stream>>>(e1fU, e1fW, fE2f);
    k_pack_big<<<(12288 + 255) / 256, 256, 0, stream>>>(e1bU, e1bW, fE2b);
    k_pack_big<<<(12288 + 255) / 256, 256, 0, stream>>>(d1fU, d1fW, fD2f);
    k_pack_big<<<(12288 + 255) / 256, 256, 0, stream>>>(d1bU, d1bW, fD2b);

    ScanP Pf, Pb;
    // 2) encoder layer 1 (xproj = embW table lookup)
    Pf.frag = fE1f; Pf.b0 = nullptr; Pf.b1 = e0fb + 384; Pf.xp = embWf; Pf.tok = X;
    Pf.out_hs = hsfw; Pf.ohs_sb = 128; Pf.ohs_st = (long long)B_ * 128;
    Pf.out_state = nullptr; Pf.os_pitch = 0; Pf.reverse = 0;
    Pb = Pf; Pb.frag = fE1b; Pb.b1 = e0bb + 384; Pb.xp = embWb; Pb.out_hs = hsbw; Pb.reverse = 1;
    k_scan_small<<<dim3(NBLK, 2), 256, 0, stream>>>(Pf, Pb);

    // 3) encoder layer 2 (fused; only final states kept -> encF)
    Pf.frag = fE2f; Pf.b0 = e1fb; Pf.b1 = e1fb + 384; Pf.xp = nullptr; Pf.tok = nullptr;
    Pf.out_hs = nullptr; Pf.ohs_sb = 0; Pf.ohs_st = 0;
    Pf.out_state = encF; Pf.os_pitch = 256; Pf.reverse = 0;
    Pb = Pf; Pb.frag = fE2b; Pb.b0 = e1bb; Pb.b1 = e1bb + 384;
    Pb.out_state = encF + 128; Pb.reverse = 1;
    k_scan_big<<<dim3(NBLK, 2), 256, 0, stream>>>(Pf, Pb, hsfw, hsbw);

    // 4) decoder layer-1 constant xproj: enc_feats @ Wd0 + b0
    k_proj_addb<<<(B_ * 384 + 255) / 256, 256, 0, stream>>>(encF, d0fW, d0fb, xpDf, B_, 256);
    k_proj_addb<<<(B_ * 384 + 255) / 256, 256, 0, stream>>>(encF, d0bW, d0bb, xpDb, B_, 256);

    // 5) decoder layer 1 (const xproj; hs reuses hsfw/hsbw)
    Pf.frag = fD1f; Pf.b0 = nullptr; Pf.b1 = d0fb + 384; Pf.xp = xpDf; Pf.tok = nullptr;
    Pf.out_hs = hsfw; Pf.ohs_sb = 128; Pf.ohs_st = (long long)B_ * 128;
    Pf.out_state = nullptr; Pf.os_pitch = 0; Pf.reverse = 0;
    Pb = Pf; Pb.frag = fD1b; Pb.b1 = d0bb + 384; Pb.xp = xpDb; Pb.out_hs = hsbw; Pb.reverse = 1;
    k_scan_small<<<dim3(NBLK, 2), 256, 0, stream>>>(Pf, Pb);

    // 6) decoder layer 2 -> dec_out (d_out[12800:]) in [B,T,256], + final states
    float* decOut = (float*)d_out + (size_t)B_ * 25;
    Pf.frag = fD2f; Pf.b0 = d1fb; Pf.b1 = d1fb + 384; Pf.xp = nullptr; Pf.tok = nullptr;
    Pf.out_hs = decOut; Pf.ohs_sb = (long long)T_ * 256; Pf.ohs_st = 256;
    Pf.out_state = decF; Pf.os_pitch = 256; Pf.reverse = 0;
    Pb = Pf; Pb.frag = fD2b; Pb.b0 = d1bb; Pb.b1 = d1bb + 384;
    Pb.out_hs = decOut + 128; Pb.out_state = decF + 128; Pb.reverse = 1;
    k_scan_big<<<dim3(NBLK, 2), 256, 0, stream>>>(Pf, Pb, hsfw, hsbw);

    // 7) classifier head -> d_out[0:B*25]
    k_head<<<B_, 128, 0, stream>>>(encF, decF, W1, b1v, W2, b2v, (float*)d_out);
}
